// global_feature_75316546503290
// MI455X (gfx1250) — compile-verified, matrix operand order checked by reading, not run on hardware
//
#include <hip/hip_runtime.h>

#define BB 8
#define LL 1024
#define DD 128
#define DI 256
#define DS 16
#define DTR 8
#define MLPD 512

typedef __attribute__((ext_vector_type(16))) _Float16 v16h;
typedef __attribute__((ext_vector_type(8)))  _Float16 v8h;
typedef __attribute__((ext_vector_type(4)))  _Float16 v4h;
typedef __attribute__((ext_vector_type(8)))  float    v8f;

__device__ __forceinline__ float sigmoidf_(float x) { return 1.0f / (1.0f + __expf(-x)); }
__device__ __forceinline__ float siluf_(float x)    { return x * sigmoidf_(x); }
__device__ __forceinline__ float geluf_(float x)    { return 0.5f * x * (1.0f + erff(x * 0.70710678118654752f)); }
__device__ __forceinline__ float softplusf_(float x){ return x > 20.0f ? x : log1pf(__expf(x)); }

// ---- async global->LDS staging (CDNA5 GLOBAL_LOAD_ASYNC_TO_LDS_B128), guarded ----
#if defined(__gfx1250__) && __has_builtin(__builtin_amdgcn_global_load_async_to_lds_b128)
#define HAVE_ASYNC_LDS 1
typedef int v4i_ __attribute__((vector_size(16)));
typedef __attribute__((address_space(1))) v4i_* gas_p;   // pointer to int4 in global AS
typedef __attribute__((address_space(3))) v4i_* las_p;   // pointer to int4 in LDS AS
__device__ __forceinline__ void async_copy16(const _Float16* g, _Float16* l) {
  __builtin_amdgcn_global_load_async_to_lds_b128(
      (gas_p)(unsigned long long)(uintptr_t)g,
      (las_p)(unsigned int)(uintptr_t)l, 0, 0);
}
__device__ __forceinline__ void async_wait0() {
#if __has_builtin(__builtin_amdgcn_s_wait_asynccnt)
  __builtin_amdgcn_s_wait_asynccnt(0);
#else
  asm volatile("s_wait_asynccnt 0x0" ::: "memory");
#endif
}
#else
#define HAVE_ASYNC_LDS 0
#endif

// ---------------- LayerNorm: one wave (32 lanes) per token, D=128 ----------------
template<bool COPY_PREX>
__global__ __launch_bounds__(256) void ln_kernel(const float* __restrict__ x,
                                                 const float* __restrict__ w,
                                                 const float* __restrict__ b,
                                                 float* __restrict__ xn,
                                                 _Float16* __restrict__ xn16,
                                                 float* __restrict__ prex) {
  const int lane  = threadIdx.x & 31;
  const int token = blockIdx.x * 8 + (threadIdx.x >> 5);
  const size_t base = (size_t)token * DD + lane * 4;
  const float4 xv = *(const float4*)(x + base);
  float s  = xv.x + xv.y + xv.z + xv.w;
  float s2 = xv.x * xv.x + xv.y * xv.y + xv.z * xv.z + xv.w * xv.w;
  for (int o = 16; o > 0; o >>= 1) { s += __shfl_xor(s, o, 32); s2 += __shfl_xor(s2, o, 32); }
  const float mean = s * (1.0f / DD);
  const float var  = s2 * (1.0f / DD) - mean * mean;
  const float rs   = rsqrtf(var + 1e-5f);
  const float4 wv = *(const float4*)(w + lane * 4);
  const float4 bv = *(const float4*)(b + lane * 4);
  float4 o4;
  o4.x = (xv.x - mean) * rs * wv.x + bv.x;
  o4.y = (xv.y - mean) * rs * wv.y + bv.y;
  o4.z = (xv.z - mean) * rs * wv.z + bv.z;
  o4.w = (xv.w - mean) * rs * wv.w + bv.w;
  *(float4*)(xn + base) = o4;
  v4h h4; h4[0] = (_Float16)o4.x; h4[1] = (_Float16)o4.y; h4[2] = (_Float16)o4.z; h4[3] = (_Float16)o4.w;
  *(v4h*)(xn16 + base) = h4;
  if (COPY_PREX) *(float4*)(prex + base) = xv;
}

// ---------------- fp32 [K x N] -> fp16 transposed [N x K] ----------------
__global__ __launch_bounds__(256) void convT_kernel(const float* __restrict__ w,
                                                    _Float16* __restrict__ wT,
                                                    int K, int N) {
  const int t = blockIdx.x * 256 + threadIdx.x;
  if (t >= N * K) return;
  const int n = t / K;
  const int k = t - n * K;
  wT[t] = (_Float16)w[(size_t)k * N + n];
}

// ---------------- c2 combined weight: A[o][i] = (i<L) ? w0+w1 : w1 ----------------
__global__ __launch_bounds__(256) void c2a_kernel(const float* __restrict__ c2w,
                                                  _Float16* __restrict__ a) {
  const int t = blockIdx.x * 256 + threadIdx.x;   // t < 1024*2048
  const int o = t >> 11;
  const int i = t & 2047;
  float v;
  if (i < LL) v = c2w[((size_t)o * LL + i) * 2 + 0] + c2w[((size_t)o * LL + i) * 2 + 1];
  else        v = c2w[((size_t)o * LL + (i - LL)) * 2 + 1];
  a[t] = (_Float16)v;
}

// ---------------- pack c2 B operand: BT[b][c][i] = (i<L)?prex[b,i,c]:x[b,i,c] ----------------
__global__ __launch_bounds__(256) void pack_c2b_kernel(const float* __restrict__ prex,
                                                       const float* __restrict__ x,
                                                       _Float16* __restrict__ out) {
  const int t = blockIdx.x * 256 + threadIdx.x;   // t < 8*128*2048
  const int i = t & 2047;
  const int c = (t >> 11) & 127;
  const int b = t >> 18;
  float v;
  if (i < LL) v = prex[((size_t)(b * LL + i)) * DD + c];
  else        v = x[((size_t)(b * LL + (i - LL))) * DD + c];
  out[t] = (_Float16)v;
}

// ---------------- causal depthwise conv (K=4) + SiLU ----------------
__global__ __launch_bounds__(256) void conv_silu_kernel(const float* __restrict__ xz,
                                                        const float* __restrict__ cw,
                                                        const float* __restrict__ cb,
                                                        float* __restrict__ xc) {
  const int t = blockIdx.x * 256 + threadIdx.x;   // t < B*L*DI
  const int d = t & 255;
  const int l = (t >> 8) & 1023;
  const int b = t >> 18;
  float acc = cb[d];
#pragma unroll
  for (int j = 0; j < 4; ++j) {
    const int pos = l - 3 + j;
    if (pos >= 0) acc += cw[d * 4 + j] * xz[((size_t)(b * LL + pos)) * (2 * DI) + d];
  }
  xc[t] = siluf_(acc);
}

// ---------------- xproj: dbl[token][n] = sum_k xc[token][k]*W[k][n], N=40 ----------------
__global__ __launch_bounds__(256) void xproj_kernel(const float* __restrict__ xc,
                                                    const float* __restrict__ pw,
                                                    float* __restrict__ dbl) {
  const int t = blockIdx.x * 256 + threadIdx.x;   // t < 8192*40
  const int token = t / 40;
  const int n = t - token * 40;
  float acc = 0.0f;
  const float* xr = xc + (size_t)token * DI;
  for (int k = 0; k < DI; ++k) acc += xr[k] * pw[k * 40 + n];
  dbl[t] = acc;
}

// ---------------- dt = softplus(dtr @ dt_w + dt_b) ----------------
__global__ __launch_bounds__(256) void dt_kernel(const float* __restrict__ dbl,
                                                 const float* __restrict__ dtw,
                                                 const float* __restrict__ dtb,
                                                 float* __restrict__ dt) {
  const int t = blockIdx.x * 256 + threadIdx.x;   // t < 8192*256
  const int d = t & 255;
  const int token = t >> 8;
  float acc = dtb[d];
#pragma unroll
  for (int r = 0; r < DTR; ++r) acc += dbl[(size_t)token * 40 + r] * dtw[r * DI + d];
  dt[t] = softplusf_(acc);
}

// ---------------- selective scan: thread per (b,d,s), 16-lane reduce over s ----------------
__global__ __launch_bounds__(256) void scan_kernel(const float* __restrict__ dt,
                                                   const float* __restrict__ xc,
                                                   const float* __restrict__ xz,
                                                   const float* __restrict__ dbl,
                                                   const float* __restrict__ A_log,
                                                   const float* __restrict__ Dskip,
                                                   _Float16* __restrict__ y16) {
  const int t = blockIdx.x * 256 + threadIdx.x;   // t < 8*256*16
  const int s = t & 15;
  const int d = (t >> 4) & 255;
  const int b = t >> 12;
  const float As  = -__expf(A_log[d * DS + s]);
  const float Dsk = Dskip[d];
  float h = 0.0f;
  for (int l = 0; l < LL; ++l) {
    const size_t base = (size_t)(b * LL + l);
    const float dtv = dt[base * DI + d];
    const float xcv = xc[base * DI + d];
    const float Bcs = dbl[base * 40 + DTR + s];
    const float Ccs = dbl[base * 40 + DTR + DS + s];
    h = __expf(dtv * As) * h + dtv * Bcs * xcv;
    float part = h * Ccs;
    part += __shfl_xor(part, 1, 32);
    part += __shfl_xor(part, 2, 32);
    part += __shfl_xor(part, 4, 32);
    part += __shfl_xor(part, 8, 32);
    if (s == 0) {
      const float zv = xz[base * (2 * DI) + DI + d];
      y16[base * DI + d] = (_Float16)((part + Dsk * xcv) * siluf_(zv));
    }
  }
}

// ---------------- WMMA GEMM: C[MxN] = epi(A16[MxK] @ BT16[NxK]^T) ----------------
// EPI: 0 plain f32 | 1 +colbias,GELU,store f16 | 2 +colbias+res1 | 3 +res1+res2 | 4 +rowbias (batched)
template<int EPI>
__global__ __launch_bounds__(256) void gemm16(const _Float16* __restrict__ A,
                                              const _Float16* __restrict__ BT,
                                              float* __restrict__ C,
                                              _Float16* __restrict__ C16,
                                              const float* __restrict__ bias,
                                              const float* __restrict__ res1,
                                              const float* __restrict__ res2,
                                              int M, int N, int K,
                                              int strideBT, int strideC) {
  __shared__ _Float16 sA[64 * 32];
  __shared__ _Float16 sB[64 * 32];
  const int tid  = threadIdx.x;
  const int lane = tid & 31;
  const int wave = tid >> 5;
  const int wm = wave >> 2;          // 0..1 (M)
  const int wn = wave & 3;           // 0..3 (N)
  const int bm = blockIdx.y * 64;
  const int bn = blockIdx.x * 64;
  const _Float16* Bb = BT + (size_t)blockIdx.z * strideBT;
  const int lrow = tid >> 2;
  const int lcol = (tid & 3) * 8;
  const int fm = lane & 15;          // M (A) or N (B) within 16-tile
  const int kh = lane >> 4;          // K half select

  v8f acc[2] = {};
  for (int k0 = 0; k0 < K; k0 += 32) {
    const _Float16* gA = A  + (size_t)(bm + lrow) * K + k0 + lcol;
    const _Float16* gB = Bb + (size_t)(bn + lrow) * K + k0 + lcol;
#if HAVE_ASYNC_LDS
    async_copy16(gA, &sA[lrow * 32 + lcol]);
    async_copy16(gB, &sB[lrow * 32 + lcol]);
#else
    *(v8h*)&sA[lrow * 32 + lcol] = *(const v8h*)gA;
    *(v8h*)&sB[lrow * 32 + lcol] = *(const v8h*)gB;
#endif
    if (k0 + 32 < K) {
      __builtin_prefetch(gA + 32, 0, 1);
      __builtin_prefetch(gB + 32, 0, 1);
    }
#if HAVE_ASYNC_LDS
    async_wait0();
#endif
    __syncthreads();
    const v8h a0lo = *(const v8h*)&sA[(wm * 32 + fm) * 32 + kh * 8];
    const v8h a0hi = *(const v8h*)&sA[(wm * 32 + fm) * 32 + 16 + kh * 8];
    const v8h a1lo = *(const v8h*)&sA[(wm * 32 + 16 + fm) * 32 + kh * 8];
    const v8h a1hi = *(const v8h*)&sA[(wm * 32 + 16 + fm) * 32 + 16 + kh * 8];
    const v8h blo  = *(const v8h*)&sB[(wn * 16 + fm) * 32 + kh * 8];
    const v8h bhi  = *(const v8h*)&sB[(wn * 16 + fm) * 32 + 16 + kh * 8];
    const v16h af0 = __builtin_shufflevector(a0lo, a0hi, 0,1,2,3,4,5,6,7,8,9,10,11,12,13,14,15);
    const v16h af1 = __builtin_shufflevector(a1lo, a1hi, 0,1,2,3,4,5,6,7,8,9,10,11,12,13,14,15);
    const v16h bf  = __builtin_shufflevector(blo,  bhi,  0,1,2,3,4,5,6,7,8,9,10,11,12,13,14,15);
    acc[0] = __builtin_amdgcn_wmma_f32_16x16x32_f16(false, af0, false, bf, (short)0, acc[0], false, false);
    acc[1] = __builtin_amdgcn_wmma_f32_16x16x32_f16(false, af1, false, bf, (short)0, acc[1], false, false);
    __syncthreads();
  }

  const int col = bn + wn * 16 + fm;
  const size_t zoff = (size_t)blockIdx.z * strideC;
#pragma unroll
  for (int sub = 0; sub < 2; ++sub) {
#pragma unroll
    for (int r = 0; r < 8; ++r) {
      const int row = bm + wm * 32 + sub * 16 + kh * 8 + r;
      const size_t idx = (size_t)row * N + col;
      float v = acc[sub][r];
      if (EPI == 1 || EPI == 2) v += bias[col];
      if (EPI == 4) v += bias[row];
      if (EPI == 1) v = geluf_(v);
      if (EPI == 2) v += res1[idx];
      if (EPI == 3) v += res1[idx] + res2[idx];
      if (EPI == 1) C16[idx] = (_Float16)v;
      else          C[zoff + idx] = v;
    }
  }
}

extern "C" void kernel_launch(void* const* d_in, const int* in_sizes, int n_in,
                              void* d_out, int out_size, void* d_ws, size_t ws_size,
                              hipStream_t stream) {
  (void)in_sizes; (void)n_in; (void)out_size;
  const float* x_in = (const float*)d_in[0];
  auto P = [&](int layer, int idx) -> const float* {
    return (const float*)d_in[1 + layer * 19 + idx];
  };

  size_t off = 0;
  auto carve = [&](size_t bytes) -> void* {
    void* p = (char*)d_ws + off;
    off = (off + bytes + 255) & ~(size_t)255;
    return p;
  };
  float*    prex  = (float*)carve((size_t)BB * LL * DD * 4);
  float*    xn    = (float*)carve((size_t)BB * LL * DD * 4);
  float*    xz    = (float*)carve((size_t)BB * LL * 2 * DI * 4);
  float*    xc    = (float*)carve((size_t)BB * LL * DI * 4);
  float*    dtb   = (float*)carve((size_t)BB * LL * DI * 4);
  float*    dbl   = (float*)carve((size_t)BB * LL * 40 * 4);
  _Float16* xn16  = (_Float16*)carve((size_t)BB * LL * DD * 2);
  _Float16* y16   = (_Float16*)carve((size_t)BB * LL * DI * 2);
  _Float16* g16   = (_Float16*)carve((size_t)BB * LL * MLPD * 2);
  _Float16* c2B   = (_Float16*)carve((size_t)BB * DD * 2 * LL * 2);
  _Float16* c2A   = (_Float16*)carve((size_t)LL * 2 * LL * 2);
  _Float16* wInT  = (_Float16*)carve((size_t)DD * 2 * DI * 2);
  _Float16* wOutT = (_Float16*)carve((size_t)DI * DD * 2);
  _Float16* wF1T  = (_Float16*)carve((size_t)DD * MLPD * 2);
  _Float16* wF2T  = (_Float16*)carve((size_t)MLPD * DD * 2);
  if (off > ws_size) return;

  float* xbuf = (float*)d_out;   // running activation buffer, final result lands here
  (void)hipMemcpyAsync(xbuf, x_in, (size_t)BB * LL * DD * 4, hipMemcpyDeviceToDevice, stream);

  for (int l = 0; l < 2; ++l) {
    // prex = x; xn = LN1(x)
    ln_kernel<true><<<1024, 256, 0, stream>>>(xbuf, P(l, 0), P(l, 1), xn, xn16, prex);

    // weight conversion (f16, transposed for B operand)
    convT_kernel<<<(DD * 2 * DI) / 256, 256, 0, stream>>>(P(l, 2),  wInT,  DD,  2 * DI);
    convT_kernel<<<(DI * DD) / 256,     256, 0, stream>>>(P(l, 10), wOutT, DI,  DD);
    convT_kernel<<<(DD * MLPD) / 256,   256, 0, stream>>>(P(l, 13), wF1T,  DD,  MLPD);
    convT_kernel<<<(MLPD * DD) / 256,   256, 0, stream>>>(P(l, 15), wF2T,  MLPD, DD);
    c2a_kernel<<<(LL * 2 * LL) / 256,   256, 0, stream>>>(P(l, 17), c2A);

    // xz = xn @ in_w   [8192x512, K=128]
    gemm16<0><<<dim3(8, 128, 1), 256, 0, stream>>>(xn16, wInT, xz, nullptr,
        nullptr, nullptr, nullptr, BB * LL, 2 * DI, DD, 0, 0);

    conv_silu_kernel<<<(BB * LL * DI) / 256, 256, 0, stream>>>(xz, P(l, 3), P(l, 4), xc);
    xproj_kernel<<<(BB * LL * 40) / 256,     256, 0, stream>>>(xc, P(l, 5), dbl);
    dt_kernel<<<(BB * LL * DI) / 256,        256, 0, stream>>>(dbl, P(l, 6), P(l, 7), dtb);
    scan_kernel<<<(BB * DI * DS) / 256,      256, 0, stream>>>(dtb, xc, xz, dbl,
        P(l, 8), P(l, 9), y16);

    // x = y @ out_w + xn + x   [8192x128, K=256]
    gemm16<3><<<dim3(2, 128, 1), 256, 0, stream>>>(y16, wOutT, xbuf, nullptr,
        nullptr, xn, xbuf, BB * LL, DD, DI, 0, 0);

    // xn = LN2(x)
    ln_kernel<false><<<1024, 256, 0, stream>>>(xbuf, P(l, 11), P(l, 12), xn, xn16, nullptr);

    // g = gelu(xn @ ff_w1 + b1) -> f16   [8192x512, K=128]
    gemm16<1><<<dim3(8, 128, 1), 256, 0, stream>>>(xn16, wF1T, nullptr, g16,
        P(l, 14), nullptr, nullptr, BB * LL, MLPD, DD, 0, 0);

    // x = g @ ff_w2 + b2 + x   [8192x128, K=512]
    gemm16<2><<<dim3(2, 128, 1), 256, 0, stream>>>(g16, wF2T, xbuf, nullptr,
        P(l, 16), xbuf, nullptr, BB * LL, DD, MLPD, 0, 0);

    // c2: x = [Wc|W1] @ [prex;x] + c2_b[o]   batched over b, [1024x128, K=2048]
    pack_c2b_kernel<<<(BB * DD * 2 * LL) / 256, 256, 0, stream>>>(prex, xbuf, c2B);
    gemm16<4><<<dim3(2, 16, 8), 256, 0, stream>>>(c2A, c2B, xbuf, nullptr,
        P(l, 18), nullptr, nullptr, LL, DD, 2 * LL, DD * 2 * LL, LL * DD);
  }
}
